// MultiHeadAttention_44178033606903
// MI455X (gfx1250) — matrix operand order repaired and checked by reading, not compiled, not run on hardware
//
#include <hip/hip_runtime.h>

typedef unsigned short u16;
typedef unsigned int u32;

typedef __attribute__((ext_vector_type(16))) __bf16 v16bf;
typedef __attribute__((ext_vector_type(8)))  float  v8f;

struct __align__(16) U4 { u32 x, y, z, w; };
struct __align__(8)  F2 { float x, y; };

union Frag {
  v16bf v;
  U4 q[2];
  u32 u[8];
  u16 s[16];
};

__device__ __forceinline__ u16 f2bf(float f) {
  union { float f; u32 u; } c; c.f = f;
  u32 r = c.u + 0x7FFFu + ((c.u >> 16) & 1u);
  return (u16)(r >> 16);
}
__device__ __forceinline__ u32 pack_bf(float lo, float hi) {
  return (u32)f2bf(lo) | ((u32)f2bf(hi) << 16);
}
__device__ __forceinline__ v8f wmma_bf16(v16bf a, v16bf b, v8f c) {
  return __builtin_amdgcn_wmma_f32_16x16x32_bf16(false, a, false, b, (short)0, c, false, false);
}

// ---------------------------------------------------------------------------
// Prep: fp32 -> bf16 conversion with optional elementwise mask.
// ---------------------------------------------------------------------------
__global__ void conv_bf16_kernel(const float* __restrict__ src,
                                 const float* __restrict__ msk,
                                 u16* __restrict__ dst, int n) {
  int i = blockIdx.x * 256 + threadIdx.x;
  if (i < n) {
    float v = src[i];
    if (msk) v *= msk[i];
    dst[i] = f2bf(v);
  }
}

// ---------------------------------------------------------------------------
// Tiled bf16 WMMA GEMM: C[M,N] = A[M,K] * B[K,N], block tile 128x64, 8 waves.
// Register-pipelined global staging; 4 back-to-back WMMAs per K-step.
// mode 0: write bf16 into (B,H,L,D) head-split layout (projections).
// mode 1: write f32 + bias into d_out (final projection).
// ---------------------------------------------------------------------------
__global__ __launch_bounds__(256) void gemm_bf16_kernel(
    const u16* __restrict__ A, const u16* __restrict__ Bm,
    int M, int N, int K, int mode,
    u16* __restrict__ outb, float* __restrict__ outf,
    const float* __restrict__ bias) {
  __shared__ u16 As[128][40];   // 128 x 32 tile, +8 pad
  __shared__ u16 Bs[64][40];    // transposed 32 x 64 tile -> [n][k]

  const int tid = threadIdx.x;
  const int w = tid >> 5, lane = tid & 31;
  const int hf = lane >> 4, l16 = lane & 15;
  const int m0 = blockIdx.x * 128, n0 = blockIdx.y * 64;

  const int ra = tid >> 1, ca = (tid & 1) * 16;
  const int rb = tid >> 3, cb = (tid & 7) * 8;
  const u16* aptr = A + (size_t)(m0 + ra) * K + ca;
  const u16* bptr = Bm + (size_t)rb * N + n0 + cb;

  // prologue: stage first K-chunk into registers
  U4 rga0 = *reinterpret_cast<const U4*>(aptr);
  U4 rga1 = *reinterpret_cast<const U4*>(aptr + 8);
  U4 rgb  = *reinterpret_cast<const U4*>(bptr);

  v8f acc[4];
  #pragma unroll
  for (int t = 0; t < 4; ++t) acc[t] = (v8f){0.f,0.f,0.f,0.f,0.f,0.f,0.f,0.f};

  const int ml = (w << 4) + l16;

  for (int k0 = 0; k0 < K; k0 += 32) {
    __syncthreads();
    *reinterpret_cast<U4*>(&As[ra][ca]) = rga0;
    *reinterpret_cast<U4*>(&As[ra][ca + 8]) = rga1;
    {
      const u16* xs = reinterpret_cast<const u16*>(&rgb);
      #pragma unroll
      for (int i = 0; i < 8; ++i) Bs[cb + i][rb] = xs[i];
    }
    __syncthreads();

    if (k0 + 32 < K) {  // software pipeline: fetch next chunk during compute
      aptr += 32;
      bptr += (size_t)32 * N;
      rga0 = *reinterpret_cast<const U4*>(aptr);
      rga1 = *reinterpret_cast<const U4*>(aptr + 8);
      rgb  = *reinterpret_cast<const U4*>(bptr);
    }

    Frag af;
    af.q[0] = *reinterpret_cast<const U4*>(&As[ml][hf * 16]);
    af.q[1] = *reinterpret_cast<const U4*>(&As[ml][hf * 16 + 8]);
    Frag bfr[4];
    #pragma unroll
    for (int nt = 0; nt < 4; ++nt) {
      int nl = nt * 16 + l16;
      bfr[nt].q[0] = *reinterpret_cast<const U4*>(&Bs[nl][hf * 16]);
      bfr[nt].q[1] = *reinterpret_cast<const U4*>(&Bs[nl][hf * 16 + 8]);
    }
    #pragma unroll
    for (int nt = 0; nt < 4; ++nt)
      acc[nt] = wmma_bf16(af.v, bfr[nt].v, acc[nt]);
  }

  #pragma unroll
  for (int nt = 0; nt < 4; ++nt) {
    #pragma unroll
    for (int r = 0; r < 8; ++r) {
      int mg = m0 + (w << 4) + r + 8 * hf;
      int ng = n0 + nt * 16 + l16;
      float vv = acc[nt][r];
      if (mode == 0) {
        int bb = mg >> 9, l = mg & 511, hh = ng >> 6, dd = ng & 63;
        outb[(((size_t)(bb * 8 + hh) * 512) + l) * 64 + dd] = f2bf(vv);
      } else {
        outf[(size_t)mg * N + ng] = vv + bias[ng];
      }
    }
  }
}

// ---------------------------------------------------------------------------
// Fused attention per (b,h, 32-query tile).
// ---------------------------------------------------------------------------
__global__ __launch_bounds__(256) void attn_kernel(
    const u16* __restrict__ qh, const u16* __restrict__ kh,
    const u16* __restrict__ vh, const u16* __restrict__ krel,
    const float* __restrict__ vrel_g, const float* __restrict__ pad_mask,
    float* __restrict__ attn_out, u16* __restrict__ ctx_bf) {
  __shared__ float att[32][516];     // logits -> attention (f32)
  __shared__ float relw[32][144];    // rel logits, then bucket weights w
  __shared__ float vrelS[129][64];   // value_rel[h]
  __shared__ float ctxs[32][64];     // attn @ v result
  __shared__ u16   qs[32][72];       // q tile bf16
  __shared__ u16   ks[64][72];       // k tile [j][d]; reused as v^T [d][j]

  const int tid = threadIdx.x;
  const int w = tid >> 5, lane = tid & 31;
  const int hf = lane >> 4, l16 = lane & 15;
  const int bh = blockIdx.y;            // b*8 + h
  const int b = bh >> 3, h = bh & 7;
  const int l0 = blockIdx.x * 32;
  const int mi = w & 1, ni = w >> 1;    // wave's 16x16 subtile position
  const int ml = mi * 16 + l16;

  { // load 32x64 q tile (bf16)
    int r = tid >> 3, c = (tid & 7) * 8;
    const U4* g = reinterpret_cast<const U4*>(qh + ((size_t)bh * 512 + l0 + r) * 64 + c);
    *reinterpret_cast<U4*>(&qs[r][c]) = g[0];
  }
  for (int i = tid; i < 129 * 64; i += 256)   // stage value_rel[h]
    vrelS[i >> 6][i & 63] = vrel_g[(size_t)h * 129 * 64 + i];
  __syncthreads();

  // hoisted, loop-invariant A-fragments (q rows of this wave's subtile)
  Frag aq[2];
  #pragma unroll
  for (int kc = 0; kc < 2; ++kc) {
    aq[kc].q[0] = *reinterpret_cast<const U4*>(&qs[ml][kc * 32 + hf * 16]);
    aq[kc].q[1] = *reinterpret_cast<const U4*>(&qs[ml][kc * 32 + hf * 16 + 8]);
  }

  // rel[l,m] = q[l] . key_rel[h,m]  (32x129, padded to 144), via WMMA
  for (int st = w; st < 18; st += 8) {
    int smi = st & 1, nt = st >> 1;
    int ncol = nt * 16 + l16;
    int sml = smi * 16 + l16;
    v8f acc = (v8f){0.f,0.f,0.f,0.f,0.f,0.f,0.f,0.f};
    #pragma unroll
    for (int kc = 0; kc < 2; ++kc) {
      Frag af, bf;
      af.q[0] = *reinterpret_cast<const U4*>(&qs[sml][kc * 32 + hf * 16]);
      af.q[1] = *reinterpret_cast<const U4*>(&qs[sml][kc * 32 + hf * 16 + 8]);
      if (ncol < 129) {
        const U4* g = reinterpret_cast<const U4*>(
            krel + ((size_t)h * 129 + ncol) * 64 + kc * 32 + hf * 16);
        bf.q[0] = g[0]; bf.q[1] = g[1];
      } else {
        bf.q[0] = U4{0,0,0,0}; bf.q[1] = U4{0,0,0,0};
      }
      acc = wmma_bf16(af.v, bf.v, acc);
    }
    #pragma unroll
    for (int r = 0; r < 8; ++r)
      relw[smi * 16 + r + 8 * hf][nt * 16 + l16] = acc[r];
  }
  __syncthreads();

  // logits = q k^T, + skewed rel, scaled, + pad mask (pipelined k tiles)
  const int rk = tid >> 2, ck = (tid & 3) * 16;
  const u16* kp = kh + ((size_t)bh * 512 + rk) * 64 + ck;
  U4 kg0 = *reinterpret_cast<const U4*>(kp);
  U4 kg1 = *reinterpret_cast<const U4*>(kp + 8);

  for (int kt = 0; kt < 8; ++kt) {
    __syncthreads();
    *reinterpret_cast<U4*>(&ks[rk][ck]) = kg0;
    *reinterpret_cast<U4*>(&ks[rk][ck + 8]) = kg1;
    __syncthreads();
    if (kt < 7) {
      kp += 64 * 64;
      kg0 = *reinterpret_cast<const U4*>(kp);
      kg1 = *reinterpret_cast<const U4*>(kp + 8);
    }
    const int jn = ni * 16 + l16;
    Frag bf0, bf1;
    bf0.q[0] = *reinterpret_cast<const U4*>(&ks[jn][hf * 16]);
    bf0.q[1] = *reinterpret_cast<const U4*>(&ks[jn][hf * 16 + 8]);
    bf1.q[0] = *reinterpret_cast<const U4*>(&ks[jn][32 + hf * 16]);
    bf1.q[1] = *reinterpret_cast<const U4*>(&ks[jn][32 + hf * 16 + 8]);
    v8f acc = (v8f){0.f,0.f,0.f,0.f,0.f,0.f,0.f,0.f};
    acc = wmma_bf16(aq[0].v, bf0.v, acc);
    acc = wmma_bf16(aq[1].v, bf1.v, acc);
    #pragma unroll
    for (int r = 0; r < 8; ++r) {
      int ll = mi * 16 + r + 8 * hf;
      int jg = kt * 64 + ni * 16 + l16;
      int lg = l0 + ll;
      int ridx = jg - lg + 64;
      ridx = ridx < 0 ? 0 : (ridx > 128 ? 128 : ridx);
      att[ll][jg] = (acc[r] + relw[ll][ridx]) * 0.125f
                    + pad_mask[b * 512 + jg] * (-1.0e9f);
    }
  }

  // prologue loads for the V tiles: overlap with softmax below
  const u16* vp = vh + ((size_t)bh * 512 + rk) * 64 + ck;
  U4 vg0 = *reinterpret_cast<const U4*>(vp);
  U4 vg1 = *reinterpret_cast<const U4*>(vp + 8);

  __syncthreads();

  // softmax per row; write attn output (f32)
  for (int rr = 0; rr < 4; ++rr) {
    int ll = w * 4 + rr, lg = l0 + ll;
    float mx = -3.4e38f;
    for (int c = lane; c < 512; c += 32) mx = fmaxf(mx, att[ll][c]);
    #pragma unroll
    for (int off = 16; off > 0; off >>= 1) mx = fmaxf(mx, __shfl_xor(mx, off, 32));
    float s = 0.f;
    for (int c = lane; c < 512; c += 32) {
      float e = __expf(att[ll][c] - mx); att[ll][c] = e; s += e;
    }
    #pragma unroll
    for (int off = 16; off > 0; off >>= 1) s += __shfl_xor(s, off, 32);
    float inv = 1.0f / s;
    float* grow = attn_out + ((size_t)bh * 512 + lg) * 512;
    for (int c = lane; c < 512; c += 32) {
      float a = att[ll][c] * inv; att[ll][c] = a; grow[c] = a;
    }
  }
  __syncthreads();

  // bucket weights w[l,m] = attn mass at clamped relative offset m
  for (int rr = 0; rr < 4; ++rr) {
    int ll = w * 4 + rr, lg = l0 + ll;
    float s0 = 0.f, s1 = 0.f;
    for (int c = lane; c <= lg - 64; c += 32) s0 += att[ll][c];
    for (int c = lg + 64 + lane; c < 512; c += 32) s1 += att[ll][c];
    #pragma unroll
    for (int off = 16; off > 0; off >>= 1) {
      s0 += __shfl_xor(s0, off, 32); s1 += __shfl_xor(s1, off, 32);
    }
    if (lane == 0) { relw[ll][0] = s0; relw[ll][128] = s1; }
    for (int m = 1 + lane; m < 128; m += 32) {
      int j = lg + m - 64;
      relw[ll][m] = (j >= 0 && j < 512) ? att[ll][j] : 0.f;
    }
  }
  __syncthreads();

  // rel-value term: acc2 = w @ value_rel[h]   (per-thread 129x8 MAC)
  float acc2[8];
  const int pl = tid >> 3, pd = (tid & 7) * 8;
  #pragma unroll
  for (int i = 0; i < 8; ++i) acc2[i] = 0.f;
  for (int m = 0; m < 129; ++m) {
    float wv = relw[pl][m];
    #pragma unroll
    for (int i = 0; i < 8; ++i) acc2[i] += wv * vrelS[m][pd + i];
  }

  // out1 = attn @ v via WMMA (A built on the fly from f32 attn in LDS)
  v8f acc1 = (v8f){0.f,0.f,0.f,0.f,0.f,0.f,0.f,0.f};
  for (int vc = 0; vc < 8; ++vc) {
    __syncthreads();
    { // scatter-store 64x64 v tile transposed -> ks as [d][j]
      const u16* a0 = reinterpret_cast<const u16*>(&vg0);
      const u16* a1 = reinterpret_cast<const u16*>(&vg1);
      #pragma unroll
      for (int i = 0; i < 8; ++i) { ks[ck + i][rk] = a0[i]; ks[ck + 8 + i][rk] = a1[i]; }
    }
    __syncthreads();
    if (vc < 7) {
      vp += 64 * 64;
      vg0 = *reinterpret_cast<const U4*>(vp);
      vg1 = *reinterpret_cast<const U4*>(vp + 8);
    }
    #pragma unroll
    for (int kc = 0; kc < 2; ++kc) {
      Frag af, bf;
      int dcol = ni * 16 + l16;
      bf.q[0] = *reinterpret_cast<const U4*>(&ks[dcol][kc * 32 + hf * 16]);
      bf.q[1] = *reinterpret_cast<const U4*>(&ks[dcol][kc * 32 + hf * 16 + 8]);
      int jb0 = vc * 64 + kc * 32 + hf * 16;
      const float* arow = att[ml];
      #pragma unroll
      for (int p = 0; p < 4; ++p) {
        F2 f2 = *reinterpret_cast<const F2*>(&arow[jb0 + 2 * p]);
        af.u[p] = pack_bf(f2.x, f2.y);
      }
      #pragma unroll
      for (int p = 0; p < 4; ++p) {
        F2 f2 = *reinterpret_cast<const F2*>(&arow[jb0 + 8 + 2 * p]);
        af.u[4 + p] = pack_bf(f2.x, f2.y);
      }
      acc1 = wmma_bf16(af.v, bf.v, acc1);
    }
  }

  __syncthreads();
  #pragma unroll
  for (int r = 0; r < 8; ++r)
    ctxs[mi * 16 + r + 8 * hf][ni * 16 + l16] = acc1[r];
  __syncthreads();

  { // merge out1 + out2, store bf16 context in (B,L,UNITS) layout
    int lg = l0 + pl;
    u32 op[4];
    #pragma unroll
    for (int p = 0; p < 4; ++p) {
      float x = ctxs[pl][pd + 2 * p]     + acc2[2 * p];
      float y = ctxs[pl][pd + 2 * p + 1] + acc2[2 * p + 1];
      op[p] = pack_bf(x, y);
    }
    U4 o{op[0], op[1], op[2], op[3]};
    *reinterpret_cast<U4*>(ctx_bf + ((size_t)(b * 512 + lg)) * 512 + h * 64 + pd) = o;
  }
}

// ---------------------------------------------------------------------------
extern "C" void kernel_launch(void* const* d_in, const int* in_sizes, int n_in,
                              void* d_out, int out_size, void* d_ws, size_t ws_size,
                              hipStream_t stream) {
  const float* v_in     = (const float*)d_in[0];
  const float* k_in     = (const float*)d_in[1];
  const float* q_in     = (const float*)d_in[2];
  const float* pad_mask = (const float*)d_in[3];
  const float* wq       = (const float*)d_in[4];
  const float* wk       = (const float*)d_in[5];
  const float* wv       = (const float*)d_in[6];
  const float* wo       = (const float*)d_in[7];
  const float* wo_bias  = (const float*)d_in[8];
  const float* mask_q   = (const float*)d_in[9];
  const float* mask_k   = (const float*)d_in[10];
  const float* mask_v   = (const float*)d_in[11];
  const float* mask_o   = (const float*)d_in[12];
  const float* key_rel  = (const float*)d_in[13];
  const float* val_rel  = (const float*)d_in[14];

  u16* ws = (u16*)d_ws;                     // offsets in bf16 elements
  u16* Wq  = ws;                            // 512x512
  u16* Wk  = ws + 262144;
  u16* Wv  = ws + 524288;
  u16* Wo  = ws + 786432;
  u16* Xq  = ws + 1048576;                  // 2048x512
  u16* Xk  = ws + 2097152;
  u16* Xv  = ws + 3145728;
  u16* Qh  = ws + 4194304;                  // (B,H,L,D)
  u16* Kh  = ws + 5242880;
  u16* Vh  = ws + 6291456;
  u16* Kr  = ws + 7340032;                  // 8x129x64
  u16* Ctx = ws + 7406080;                  // (B,L,512)

  float* out_f  = (float*)d_out;            // (B,L,512)
  float* attn_f = out_f + 1048576;          // (B,H,L,L)

  const int W2 = 262144, X2 = 1048576, R2 = 8 * 129 * 64;
  conv_bf16_kernel<<<(W2 + 255) / 256, 256, 0, stream>>>(wq, mask_q, Wq, W2);
  conv_bf16_kernel<<<(W2 + 255) / 256, 256, 0, stream>>>(wk, mask_k, Wk, W2);
  conv_bf16_kernel<<<(W2 + 255) / 256, 256, 0, stream>>>(wv, mask_v, Wv, W2);
  conv_bf16_kernel<<<(W2 + 255) / 256, 256, 0, stream>>>(wo, mask_o, Wo, W2);
  conv_bf16_kernel<<<(X2 + 255) / 256, 256, 0, stream>>>(q_in, nullptr, Xq, X2);
  conv_bf16_kernel<<<(X2 + 255) / 256, 256, 0, stream>>>(k_in, nullptr, Xk, X2);
  conv_bf16_kernel<<<(X2 + 255) / 256, 256, 0, stream>>>(v_in, nullptr, Xv, X2);
  conv_bf16_kernel<<<(R2 + 255) / 256, 256, 0, stream>>>(key_rel, nullptr, Kr, R2);

  dim3 gg(16, 8);
  gemm_bf16_kernel<<<gg, 256, 0, stream>>>(Xq, Wq, 2048, 512, 512, 0, Qh, nullptr, nullptr);
  gemm_bf16_kernel<<<gg, 256, 0, stream>>>(Xk, Wk, 2048, 512, 512, 0, Kh, nullptr, nullptr);
  gemm_bf16_kernel<<<gg, 256, 0, stream>>>(Xv, Wv, 2048, 512, 512, 0, Vh, nullptr, nullptr);

  attn_kernel<<<dim3(16, 32), 256, 0, stream>>>(Qh, Kh, Vh, Kr, val_rel, pad_mask,
                                                attn_f, Ctx);

  gemm_bf16_kernel<<<gg, 256, 0, stream>>>(Ctx, Wo, 2048, 512, 512, 1, nullptr, out_f,
                                           wo_bias);
}
